// PhysNet_89378269429836
// MI455X (gfx1250) — matrix operand order repaired and checked by reading, not compiled, not run on hardware
//
#include <hip/hip_runtime.h>
#include <hip/hip_bf16.h>

// ---------------- problem constants ----------------
constexpr int NATOMS = 3840;
constexpr int NMOL   = 48;
constexpr int NPER   = 80;
constexpr int FDIM   = 128;
constexpr int KR     = 64;     // rbf count
constexpr int NBLK   = 5;
#define SR_CUT  10.0f
#define KEHALF  7.199822675975274f
#define LN2F    0.69314718055994531f

// ---------------- WMMA types ----------------
typedef __attribute__((ext_vector_type(16))) __bf16 bf16x16;
typedef __attribute__((ext_vector_type(8)))  float  f32x8;

__device__ __forceinline__ float ssp_f(float x) {
    // shifted softplus, numerically stable
    return fmaxf(x, 0.f) + __logf(1.f + __expf(-fabsf(x))) - LN2F;
}

// A-fragment (16x32 bf16), source row-major [M][K] in LDS, ld = K leading dim.
// lane l<16 : row mBase+l, K chunks {0..7, 16..23}; lane>=16: chunks {8..15, 24..31}.
__device__ __forceinline__ bf16x16 load_afrag(const __bf16* base, int ld,
                                              int mBase, int kBase, int lane) {
    int m  = mBase + (lane & 15);
    int k0 = kBase + ((lane >> 4) << 3);
    const __bf16* p = base + m * ld + k0;
    union { bf16x16 v; uint4 q[2]; } u;
    u.q[0] = *(const uint4*)(p);
    u.q[1] = *(const uint4*)(p + 16);
    return u.v;
}

// B-fragment (32x16 bf16) from TRANSPOSED LDS layout [N][K], ld = K leading dim.
// lane l holds column nBase+(l&15); K chunks {0..7, 16..23} (l<16) or {8..15, 24..31} (l>=16).
__device__ __forceinline__ bf16x16 load_bfragT(const __bf16* base, int ld,
                                               int kBase, int nBase, int lane) {
    int n  = nBase + (lane & 15);
    int k0 = kBase + ((lane >> 4) << 3);
    const __bf16* p = base + n * ld + k0;
    union { bf16x16 v; uint4 q[2]; } u;
    u.q[0] = *(const uint4*)(p);
    u.q[1] = *(const uint4*)(p + 16);
    return u.v;
}

// ---------------- kernel 0: embedding gather + Ea/Qa init ----------------
__global__ void physnet_embed(const int* __restrict__ Za, const float* __restrict__ emb,
                              float* __restrict__ x, float* __restrict__ Ea,
                              float* __restrict__ Qa) {
    int i = blockIdx.x * 256 + threadIdx.x;
    if (i < NATOMS * FDIM) {
        int n = i >> 7, f = i & 127;
        x[i] = emb[Za[n] * FDIM + f];
        if (f == 0) { Ea[n] = 0.f; Qa[n] = 0.f; }
    }
}

// ---------------- generic 64x128 WMMA GEMM: out = op( A'W + bias + s*add ) ----------------
// A [nRows,128] fp32 (preAct => shifted-softplus while staging), W [128,128] fp32 row-major.
__global__ __launch_bounds__(128) void physnet_gemm(
    const float* __restrict__ A, const float* __restrict__ W,
    const float* __restrict__ bias, const float* __restrict__ addSrc,
    const float* __restrict__ scaleVec, float* __restrict__ outF,
    __bf16* __restrict__ outB, int nRows, int preAct, int postAct) {
    __shared__ __align__(16) __bf16 sA[64 * FDIM];     // 16 KB, [row][k]
    __shared__ __align__(16) __bf16 sWT[FDIM * FDIM];  // 32 KB, [n][k] (transposed)
    __shared__ float sBias[FDIM];

    const int tid = threadIdx.x, lane = tid & 31, wave = tid >> 5;
    const int rowBase = blockIdx.x * 64;

    // warm L2/L0 for the epilogue residual tile while we do LDS staging + WMMA
    if (addSrc)
        __builtin_prefetch((const char*)(addSrc + (size_t)rowBase * FDIM) + tid * 256, 0, 1);

    for (int idx = tid; idx < 64 * FDIM; idx += 128) {
        int r = idx >> 7, c = idx & 127;
        int gr = rowBase + r;
        float v = (gr < nRows) ? A[gr * FDIM + c] : 0.f;
        if (preAct) v = ssp_f(v);
        sA[idx] = (__bf16)v;
    }
    for (int idx = tid; idx < FDIM * FDIM; idx += 128) {
        int k = idx >> 7, n = idx & 127;
        sWT[n * FDIM + k] = (__bf16)W[idx];
    }
    if (tid < FDIM) sBias[tid] = bias ? bias[tid] : 0.f;
    __syncthreads();

    const int mBase = wave * 16;
    f32x8 acc[8];
#pragma unroll
    for (int nt = 0; nt < 8; ++nt) acc[nt] = f32x8{0.f,0.f,0.f,0.f,0.f,0.f,0.f,0.f};

    // hoist all A-fragments; per K-step preload all B-fragments so the 8 WMMAs
    // issue back-to-back instead of one s_wait_dscnt per matrix op.
    bf16x16 afrag[4];
#pragma unroll
    for (int kt = 0; kt < 4; ++kt)
        afrag[kt] = load_afrag(sA, FDIM, mBase, kt * 32, lane);

#pragma unroll
    for (int kt = 0; kt < 4; ++kt) {
        bf16x16 bfrag[8];
#pragma unroll
        for (int nt = 0; nt < 8; ++nt)
            bfrag[nt] = load_bfragT(sWT, FDIM, kt * 32, nt * 16, lane);
#pragma unroll
        for (int nt = 0; nt < 8; ++nt)
            acc[nt] = __builtin_amdgcn_wmma_f32_16x16x32_bf16(
                false, afrag[kt], false, bfrag[nt], (short)0, acc[nt], false, false);
    }

    const int rOff = (lane >> 4) << 3;  // D layout: lanes 16-31 hold M = r+8
#pragma unroll
    for (int nt = 0; nt < 8; ++nt) {
        int col = nt * 16 + (lane & 15);
#pragma unroll
        for (int r = 0; r < 8; ++r) {
            int row = rowBase + mBase + r + rOff;
            if (row < nRows) {
                float v = acc[nt][r] + sBias[col];
                if (addSrc) {
                    float a0 = addSrc[row * FDIM + col];
                    v += scaleVec ? scaleVec[col] * a0 : a0;
                }
                if (postAct) v = ssp_f(v);
                if (outF) outF[row * FDIM + col] = v;
                if (outB) outB[row * FDIM + col] = (__bf16)v;
            }
        }
    }
}

// ---------------- message pass: m_i += sum_k Wrbf[k,f] * (R_i^T @ mj)[k,f] ----------------
// One 2-wave workgroup per (molecule, group of 8 atoms); each wave owns 4 atoms.
__global__ __launch_bounds__(64) void physnet_msg(
    const float* __restrict__ Ra, const __bf16* __restrict__ mjB,
    const float* __restrict__ Wrbf, const float* __restrict__ cent,
    const float* __restrict__ wid, float* __restrict__ M) {
    __shared__ __align__(16) __bf16 sMT[FDIM * 96];   // 24 KB, [f][j] transposed, j padded to 96
    __shared__ __align__(16) __bf16 sR[2][KR * 96];   // 24 KB, per-wave A = R^T [k][j]
    __shared__ float sPos[NPER * 3];
    __shared__ float sCent[KR], sWid[KR];
    __shared__ float sED[2][96], sCut[2][96];

    const int tid = threadIdx.x, lane = tid & 31, wave = tid >> 5;
    const int mol = blockIdx.x, agrp = blockIdx.y;
    const int aBase = mol * NPER;

    for (int idx = tid; idx < NPER * 3; idx += 64) sPos[idx] = Ra[aBase * 3 + idx];
    if (tid < KR) { sCent[tid] = cent[tid]; sWid[tid] = wid[tid]; }
    for (int idx = tid; idx < 96 * FDIM; idx += 64) {
        int j = idx >> 7, f = idx & 127;
        __bf16 v = (j < NPER) ? mjB[(aBase + j) * FDIM + f] : (__bf16)0.f;
        sMT[f * 96 + j] = v;
    }
    __syncthreads();

    const int rOff = (lane >> 4) << 3;
    for (int q = 0; q < 4; ++q) {
        const int ai = agrp * 8 + wave * 4 + q;  // atom 0..79 within molecule
        const float xi = sPos[ai*3+0], yi = sPos[ai*3+1], zi = sPos[ai*3+2];

        // per-j distance terms (intra-wave LDS; DS ops are in-order per wave)
        for (int j = lane; j < 96; j += 32) {
            float ed = 1.f, cut = 0.f;
            if (j < NPER && j != ai) {
                float dx = sPos[j*3+0]-xi, dy = sPos[j*3+1]-yi, dz = sPos[j*3+2]-zi;
                float d = sqrtf(fmaxf(dx*dx + dy*dy + dz*dz, 0.f));
                ed = __expf(-d);
                if (d < SR_CUT) {
                    float xr = d * (1.f / SR_CUT);
                    float x2 = xr*xr, x3 = x2*xr, x4 = x3*xr, x5 = x4*xr;
                    cut = 1.f - 6.f*x5 + 15.f*x4 - 10.f*x3;
                }
            }
            sED[wave][j] = ed;
            sCut[wave][j] = cut;
        }
        // stage R^T [k][j] as bf16 (A matrix)
        __bf16* R = sR[wave];
        for (int t = lane; t < KR * 96; t += 32) {
            int k = t / 96, j = t - k * 96;
            float diff = sED[wave][j] - sCent[k];
            float v = sCut[wave][j] * __expf(-sWid[k] * diff * diff);
            R[t] = (__bf16)v;
        }

        // S = R^T[64x96] @ mj[96x128]; fold Wrbf contraction into epilogue.
        // Hoist all 12 A-fragments; per N-tile preload the 3 B-fragments so the
        // 12-WMMA chain issues without per-op LDS waits.
        bf16x16 afrag[12];
#pragma unroll
        for (int mt = 0; mt < 4; ++mt)
#pragma unroll
            for (int kt = 0; kt < 3; ++kt)
                afrag[mt * 3 + kt] = load_afrag(R, 96, mt * 16, kt * 32, lane);

#pragma unroll
        for (int nt = 0; nt < 8; ++nt) {
            bf16x16 bfrag[3];
#pragma unroll
            for (int kt = 0; kt < 3; ++kt)
                bfrag[kt] = load_bfragT(sMT, 96, kt * 32, nt * 16, lane);

            float partial = 0.f;
            const int col = nt * 16 + (lane & 15);
#pragma unroll
            for (int mt = 0; mt < 4; ++mt) {
                f32x8 acc = f32x8{0.f,0.f,0.f,0.f,0.f,0.f,0.f,0.f};
#pragma unroll
                for (int kt = 0; kt < 3; ++kt)
                    acc = __builtin_amdgcn_wmma_f32_16x16x32_bf16(
                        false, afrag[mt * 3 + kt], false, bfrag[kt], (short)0, acc, false, false);
#pragma unroll
                for (int r = 0; r < 8; ++r) {
                    int k = mt * 16 + r + rOff;
                    partial += acc[r] * Wrbf[k * FDIM + col];  // fp32 weights via L2
                }
            }
            partial += __shfl_xor(partial, 16);
            if (lane < 16) M[(aBase + ai) * FDIM + col] += partial;  // wave owns atom -> no atomics
        }
    }
}

// ---------------- per-block output head: Ea/Qa += ssp(o) @ Wout + bout ----------------
__global__ __launch_bounds__(128) void physnet_out(
    const float* __restrict__ O, const float* __restrict__ Wout,
    const float* __restrict__ bout, float* __restrict__ Ea, float* __restrict__ Qa) {
    const int lane = threadIdx.x & 31, wave = threadIdx.x >> 5;
    const int n = blockIdx.x * 4 + wave;
    float e = 0.f, qv = 0.f;
    for (int f = lane; f < FDIM; f += 32) {
        float v = ssp_f(O[n * FDIM + f]);
        e  += v * Wout[f * 2 + 0];
        qv += v * Wout[f * 2 + 1];
    }
    for (int off = 16; off; off >>= 1) { e += __shfl_xor(e, off); qv += __shfl_xor(qv, off); }
    if (lane == 0) { Ea[n] += e + bout[0]; Qa[n] += qv + bout[1]; }
}

// ---------------- scale/shift + charge conservation (per molecule) ----------------
__global__ __launch_bounds__(128) void physnet_charge(
    const int* __restrict__ Za, const float* __restrict__ Escale,
    const float* __restrict__ Eshift, const float* __restrict__ Qscale,
    const float* __restrict__ Qshift, float* __restrict__ Ea, float* __restrict__ Qa) {
    __shared__ float sQ[NPER];
    __shared__ float sSum;
    const int mol = blockIdx.x, t = threadIdx.x;
    if (t < NPER) {
        int n = mol * NPER + t;
        int z = Za[n];
        Ea[n] = Escale[z] * Ea[n] + Eshift[z];
        float qv = Qscale[z] * Qa[n] + Qshift[z];
        sQ[t] = qv;
        Qa[n] = qv;
    }
    __syncthreads();
    if (t == 0) {
        float s = 0.f;
        for (int k = 0; k < NPER; ++k) s += sQ[k];
        sSum = s;
    }
    __syncthreads();
    if (t < NPER) Qa[mol * NPER + t] -= sSum * (1.f / (float)NPER);
}

// ---------------- shielded electrostatics + molecule energy reduction ----------------
__global__ __launch_bounds__(256) void physnet_elec(
    const float* __restrict__ Ra, const float* __restrict__ Ea,
    const float* __restrict__ Qa, float* __restrict__ out) {
    __shared__ float sP[NPER * 3];
    __shared__ float sQ[NPER];
    __shared__ float red[256];
    const int mol = blockIdx.x, t = threadIdx.x;
    for (int idx = t; idx < NPER * 3; idx += 256) sP[idx] = Ra[mol * NPER * 3 + idx];
    if (t < NPER) sQ[t] = Qa[mol * NPER + t];
    __syncthreads();
    float acc = 0.f;
    for (int p = t; p < NPER * NPER; p += 256) {
        int i = p / NPER, j = p - i * NPER;
        if (i != j) {
            float dx = sP[i*3+0]-sP[j*3+0], dy = sP[i*3+1]-sP[j*3+1], dz = sP[i*3+2]-sP[j*3+2];
            float d2 = dx*dx + dy*dy + dz*dz;
            float d  = sqrtf(d2);
            float ds = sqrtf(d2 + 1.f);
            float sw = 1.f;
            if (d < 0.5f * SR_CUT) {
                float xs = d * (2.f / SR_CUT);
                float x2 = xs*xs, x3 = x2*xs, x4 = x3*xs, x5 = x4*xs;
                sw = 6.f*x5 - 15.f*x4 + 10.f*x3;
            }
            acc += KEHALF * sQ[i] * sQ[j] * ((1.f - sw) / ds + sw / d);
        }
    }
    if (t < NPER) acc += Ea[mol * NPER + t];
    red[t] = acc;
    __syncthreads();
    for (int s = 128; s; s >>= 1) {
        if (t < s) red[t] += red[t + s];
        __syncthreads();
    }
    if (t == 0) out[mol] = red[0];
}

// ---------------- host orchestration ----------------
extern "C" void kernel_launch(void* const* d_in, const int* in_sizes, int n_in,
                              void* d_out, int out_size, void* d_ws, size_t ws_size,
                              hipStream_t stream) {
    (void)in_sizes; (void)n_in; (void)out_size; (void)ws_size;
    const int*   Za     = (const int*)  d_in[0];
    const float* Ra     = (const float*)d_in[1];
    // d_in[2] idx_i, d_in[3] idx_j, d_in[4] batch_seg: edge structure is implicit (all
    // intra-molecule ordered pairs, i-major) and exploited directly.
    const float* emb    = (const float*)d_in[5];
    const float* cent   = (const float*)d_in[6];
    const float* wid    = (const float*)d_in[7];
    const float* Wrbf   = (const float*)d_in[8];
    const float* Wi     = (const float*)d_in[9];
    const float* bi     = (const float*)d_in[10];
    const float* Wj     = (const float*)d_in[11];
    const float* bj     = (const float*)d_in[12];
    const float* resIW  = (const float*)d_in[13];
    const float* resIb  = (const float*)d_in[14];
    const float* Wd     = (const float*)d_in[15];
    const float* bd     = (const float*)d_in[16];
    const float* uVec   = (const float*)d_in[17];
    const float* resAW  = (const float*)d_in[18];
    const float* resAb  = (const float*)d_in[19];
    const float* resOW  = (const float*)d_in[20];
    const float* resOb  = (const float*)d_in[21];
    const float* Wout   = (const float*)d_in[22];
    const float* bout   = (const float*)d_in[23];
    const float* Escale = (const float*)d_in[24];
    const float* Eshift = (const float*)d_in[25];
    const float* Qscale = (const float*)d_in[26];
    const float* Qshift = (const float*)d_in[27];
    float* out = (float*)d_out;

    // workspace layout (~8.9 MB)
    char* w = (char*)d_ws;
    const size_t featB = (size_t)NATOMS * FDIM * sizeof(float);
    float*  x   = (float*)w;  w += featB;
    float*  bm  = (float*)w;  w += featB;   // m accumulator
    float*  bt  = (float*)w;  w += featB;   // residual temp
    float*  bo  = (float*)w;  w += featB;   // output-block temp
    __bf16* mjB = (__bf16*)w; w += (size_t)NATOMS * FDIM * sizeof(__bf16);
    float*  Ea  = (float*)w;  w += (size_t)NATOMS * sizeof(float);
    float*  Qa  = (float*)w;

    const int FF = FDIM * FDIM;
    dim3 gGemm(NATOMS / 64), bGemm(128);

    physnet_embed<<<(NATOMS * FDIM + 255) / 256, 256, 0, stream>>>(Za, emb, x, Ea, Qa);

    for (int blk = 0; blk < NBLK; ++blk) {
        // mi -> bm ; mj -> bf16 table
        physnet_gemm<<<gGemm, bGemm, 0, stream>>>(x, Wi + blk * FF, bi + blk * FDIM,
            nullptr, nullptr, bm, nullptr, NATOMS, 1, 1);
        physnet_gemm<<<gGemm, bGemm, 0, stream>>>(x, Wj + blk * FF, bj + blk * FDIM,
            nullptr, nullptr, nullptr, mjB, NATOMS, 1, 1);
        // rbf-gated message aggregation into bm (WMMA per atom, no atomics)
        physnet_msg<<<dim3(NMOL, NPER / 8), 64, 0, stream>>>(
            Ra, mjB, Wrbf + blk * KR * FDIM, cent, wid, bm);
        // interaction residual stack (NRI = 2)
        for (int r = 0; r < 2; ++r) {
            const float* W0 = resIW + (size_t)((blk * 2 + r) * 2 + 0) * FF;
            const float* b0 = resIb + (size_t)((blk * 2 + r) * 2 + 0) * FDIM;
            const float* W1 = resIW + (size_t)((blk * 2 + r) * 2 + 1) * FF;
            const float* b1 = resIb + (size_t)((blk * 2 + r) * 2 + 1) * FDIM;
            physnet_gemm<<<gGemm, bGemm, 0, stream>>>(bm, W0, b0, nullptr, nullptr,
                bt, nullptr, NATOMS, 1, 1);
            physnet_gemm<<<gGemm, bGemm, 0, stream>>>(bt, W1, b1, bm, nullptr,
                bm, nullptr, NATOMS, 0, 0);
        }
        // x = u*x + ssp(m) @ Wd + bd
        physnet_gemm<<<gGemm, bGemm, 0, stream>>>(bm, Wd + blk * FF, bd + blk * FDIM,
            x, uVec + blk * FDIM, x, nullptr, NATOMS, 1, 0);
        // atomic residual stack (NRA = 2)
        for (int r = 0; r < 2; ++r) {
            const float* W0 = resAW + (size_t)((blk * 2 + r) * 2 + 0) * FF;
            const float* b0 = resAb + (size_t)((blk * 2 + r) * 2 + 0) * FDIM;
            const float* W1 = resAW + (size_t)((blk * 2 + r) * 2 + 1) * FF;
            const float* b1 = resAb + (size_t)((blk * 2 + r) * 2 + 1) * FDIM;
            physnet_gemm<<<gGemm, bGemm, 0, stream>>>(x, W0, b0, nullptr, nullptr,
                bt, nullptr, NATOMS, 1, 1);
            physnet_gemm<<<gGemm, bGemm, 0, stream>>>(bt, W1, b1, x, nullptr,
                x, nullptr, NATOMS, 0, 0);
        }
        // output block residual (NRO = 1): o = x + ssp(ssp(x)@W0+b0)@W1 + b1
        {
            const float* W0 = resOW + (size_t)(blk * 2 + 0) * FF;
            const float* b0 = resOb + (size_t)(blk * 2 + 0) * FDIM;
            const float* W1 = resOW + (size_t)(blk * 2 + 1) * FF;
            const float* b1 = resOb + (size_t)(blk * 2 + 1) * FDIM;
            physnet_gemm<<<gGemm, bGemm, 0, stream>>>(x, W0, b0, nullptr, nullptr,
                bt, nullptr, NATOMS, 1, 1);
            physnet_gemm<<<gGemm, bGemm, 0, stream>>>(bt, W1, b1, x, nullptr,
                bo, nullptr, NATOMS, 0, 0);
        }
        physnet_out<<<NATOMS / 4, 128, 0, stream>>>(bo, Wout + blk * FDIM * 2,
                                                    bout + blk * 2, Ea, Qa);
    }

    physnet_charge<<<NMOL, 128, 0, stream>>>(Za, Escale, Eshift, Qscale, Qshift, Ea, Qa);
    physnet_elec<<<NMOL, 256, 0, stream>>>(Ra, Ea, Qa, out);
}
